// QuantizedLinear_21517786153197
// MI455X (gfx1250) — matrix operand order repaired and checked by reading, not compiled, not run on hardware
//
#include <hip/hip_runtime.h>

typedef __bf16 bf16_t;
typedef __attribute__((ext_vector_type(16))) __bf16 v16bf;
typedef __attribute__((ext_vector_type(8)))  __bf16 v8bf;
typedef __attribute__((ext_vector_type(4)))  __bf16 v4bf;
typedef __attribute__((ext_vector_type(8)))  float  v8f;
typedef __attribute__((ext_vector_type(4)))  int    v4i;

#define M_DIM 11008
#define K_DIM 4096
#define GROUP 128
#define KH (K_DIM / 2)          // packed int32 elements per weight row
#define NGRP (K_DIM / GROUP)    // 32 scale groups per row
#define SPLITS 8
#define GRP_PER_SPLIT (NGRP / SPLITS)   // 4 groups (512 k) per wave

// Block = 256 threads = 8 waves = 1 m-tile x 8 k-splits.
// Each wave: one 16-wide m-tile, two 16-row x-tiles (acc0/acc1), 1/8 of K.
__global__ void __launch_bounds__(256) qlin_wmma_kernel(
    const bf16_t* __restrict__ x,       // [32, 4096] bf16
    const int*    __restrict__ wp,      // [11008, 2048] int32 (byte per elem)
    const bf16_t* __restrict__ scales,  // [11008, 32] bf16
    const bf16_t* __restrict__ bias,    // [11008] bf16
    bf16_t*       __restrict__ out)     // [32, 11008] bf16
{
  __shared__ float red[SPLITS * 512];   // 16 KB: [split][xrow(32)][col(16)]

  const int lane  = threadIdx.x & 31;
  const int split = threadIdx.x >> 5;   // wave id = k-split, 0..7
  const int tile  = blockIdx.x;         // 0..687
  const int m_base = tile * 16;
  const int col   = lane & 15;          // column within tile / x-row within tile
  const int g     = lane >> 4;          // lane group (K striping)

  const int m = m_base + col;                         // weight row / output column
  const bf16_t* xr0 = x + (size_t)col * K_DIM;        // A0 source row
  const bf16_t* xr1 = x + (size_t)(col + 16) * K_DIM; // A1 source row
  const int* wrow = wp + (size_t)m * KH;

  // the 4 group scales of this wave's K-split: contiguous 8 B, one load
  const v4bf svec = *(const v4bf*)(scales + (size_t)m * NGRP + split * GRP_PER_SPLIT);

  v8f acc0 = {};
  v8f acc1 = {};

  const int kbase = split * (K_DIM / SPLITS);   // split * 512

#pragma unroll
  for (int gi = 0; gi < GRP_PER_SPLIT; ++gi) {
    const float s  = (float)svec[gi];
    const float sb = -8.0f * s;                 // (n - 8) * s = n*s + (-8s)
#pragma unroll
    for (int t = 0; t < GROUP / 32; ++t) {
      const int kb = kbase + gi * GROUP + t * 32;

      // --- A fragments (x, bf16). ISA 16-bit A layout:
      // lane group g holds K = {kb+8g .. +7} in v0..3 and {kb+16+8g .. +7} in v4..7.
      v8bf a0lo = *(const v8bf*)(xr0 + kb + g * 8);
      v8bf a0hi = *(const v8bf*)(xr0 + kb + 16 + g * 8);
      v8bf a1lo = *(const v8bf*)(xr1 + kb + g * 8);
      v8bf a1hi = *(const v8bf*)(xr1 + kb + 16 + g * 8);
      v16bf a0 = __builtin_shufflevector(a0lo, a0hi,
                   0,1,2,3,4,5,6,7,8,9,10,11,12,13,14,15);
      v16bf a1 = __builtin_shufflevector(a1lo, a1hi,
                   0,1,2,3,4,5,6,7,8,9,10,11,12,13,14,15);

      // --- B fragment: dequantized weights. Lane = column (weight row m),
      // K slices {kb + 8g .. +7} and {kb + 16 + 8g .. +7} = 2x 4 packed ints = 2x b128 loads.
      const int* wptr = wrow + (kb >> 1) + g * 4;
      v4i w0 = *(const v4i*)wptr;
      v4i w1 = *(const v4i*)(wptr + 8);
      v16bf bfrag;
#pragma unroll
      for (int i = 0; i < 4; ++i) {
        const int v0i = w0[i];
        const int v1i = w1[i];
        // packed byte is 0..255 -> (v >> 4) is already the high nibble, no mask
        bfrag[2*i+0]   = (bf16_t)__builtin_fmaf((float)(v0i & 15), s, sb);
        bfrag[2*i+1]   = (bf16_t)__builtin_fmaf((float)(v0i >> 4), s, sb);
        bfrag[8+2*i+0] = (bf16_t)__builtin_fmaf((float)(v1i & 15), s, sb);
        bfrag[8+2*i+1] = (bf16_t)__builtin_fmaf((float)(v1i >> 4), s, sb);
      }

      // D = A(16x32) x B(32x16) + C, one B fragment feeds both x-row tiles
      acc0 = __builtin_amdgcn_wmma_f32_16x16x32_bf16(
                 false, a0, false, bfrag, (short)0, acc0, false, false);
      acc1 = __builtin_amdgcn_wmma_f32_16x16x32_bf16(
                 false, a1, false, bfrag, (short)0, acc1, false, false);
    }
  }

  // --- stash partial sums: D layout = VGPR r -> xrow (r + 8g), lane&15 -> col
  float* slab = &red[split * 512];
#pragma unroll
  for (int r = 0; r < 8; ++r) {
    const int xrow = r + g * 8;
    slab[xrow * 16 + col]        = acc0[r];
    slab[(xrow + 16) * 16 + col] = acc1[r];
  }
  __syncthreads();

  // --- reduce 8 k-splits, add bias, emit bf16 (512 outputs per block)
#pragma unroll
  for (int k = 0; k < 2; ++k) {
    const int o    = (int)threadIdx.x + k * 256;
    const int xrow = o >> 4;
    const int oc   = o & 15;
    const float* base = &red[xrow * 16 + oc];
    float sum = 0.0f;
#pragma unroll
    for (int j = 0; j < SPLITS; ++j) sum += base[j * 512];
    const int mcol = blockIdx.x * 16 + oc;
    sum += (float)bias[mcol];
    out[(size_t)xrow * M_DIM + mcol] = (bf16_t)sum;
  }
}

extern "C" void kernel_launch(void* const* d_in, const int* in_sizes, int n_in,
                              void* d_out, int out_size, void* d_ws, size_t ws_size,
                              hipStream_t stream) {
  (void)in_sizes; (void)n_in; (void)out_size; (void)d_ws; (void)ws_size;
  const bf16_t* x      = (const bf16_t*)d_in[0];
  const int*    wpack  = (const int*)d_in[1];
  const bf16_t* scales = (const bf16_t*)d_in[2];
  const bf16_t* bias   = (const bf16_t*)d_in[3];
  bf16_t*       out    = (bf16_t*)d_out;

  // 688 m-tiles x 8 k-splits = 5504 waves; 8 waves/block -> 688 blocks
  qlin_wmma_kernel<<<dim3(688), dim3(256), 0, stream>>>(x, wpack, scales, bias, out);
}
